// Light11_17197049053620
// MI455X (gfx1250) — compile-verified, matrix operand order checked by reading, not run on hardware
//
#include <hip/hip_runtime.h>

typedef __attribute__((ext_vector_type(16))) _Float16 v16h;
typedef __attribute__((ext_vector_type(8)))  float    v8f;

#define WMMA_F16 __builtin_amdgcn_wmma_f32_16x16x32_f16

__device__ __forceinline__ float prelu_f(float v, float a) { return v > 0.f ? v : a * v; }

constexpr int clog2(int v) { return v <= 1 ? 0 : 1 + clog2(v >> 1); }
static inline int divUp(int a, int b) { return (a + b - 1) / b; }

// ---------------------------------------------------------------------------
// head: direct 3x3 conv, Cin=1 -> Cout (pow2), pad=1, bias + PReLU.
// ---------------------------------------------------------------------------
template <int Coutsh, int Hsh, int Wsh>
__global__ void k_head(const float* __restrict__ x, const float* __restrict__ w,
                       const float* __restrict__ b, const float* __restrict__ a,
                       float* __restrict__ out, int NB)
{
    constexpr int H = 1 << Hsh, W = 1 << Wsh;
    int idx = blockIdx.x * blockDim.x + threadIdx.x;
    int total = NB << (Coutsh + Hsh + Wsh);
    if (idx >= total) return;
    int xw = idx & (W - 1);
    int t  = idx >> Wsh;
    int y  = t & (H - 1);
    t >>= Hsh;
    int co = t & ((1 << Coutsh) - 1);
    int nb = t >> Coutsh;
    const float* src = x + ((size_t)nb << (Hsh + Wsh));
    float acc = b[co];
#pragma unroll
    for (int ky = 0; ky < 3; ++ky) {
#pragma unroll
        for (int kx = 0; kx < 3; ++kx) {
            int yy = y + ky - 1, xx = xw + kx - 1;
            bool inb = ((unsigned)yy < (unsigned)H) && ((unsigned)xx < (unsigned)W);
            int yyc = inb ? yy : y;            // always-valid address
            int xxc = inb ? xx : xw;
            float v = src[(yyc << Wsh) + xxc]; // unconditional load
            acc += w[co * 9 + ky * 3 + kx] * (inb ? v : 0.f);
        }
    }
    out[idx] = prelu_f(acc, a[co]);
}

// ---------------------------------------------------------------------------
// Grouped 1x1 conv as WMMA GEMM (groups=4), fully static geometry.
// M = NB<<HWsh pixels (mult of 16), per-group K=Kdim, N=Ndim (pow2 <= 32).
// Lane-half K offset folded into the base -> per-h offsets are constants.
// ---------------------------------------------------------------------------
template <bool HAS_RES, int HWsh, int Kdim, int Ndim>
__global__ void k_pw_wmma(const float* __restrict__ in, int inC0, int inCtot,
                          const float* __restrict__ w, const float* __restrict__ bias,
                          const float* __restrict__ alpha,
                          const float* __restrict__ res, int resC0, int resCtot,
                          float* __restrict__ out, int outC0, int outCtot, int NB)
{
    constexpr int HW = 1 << HWsh;
    constexpr int ntnsh = (Ndim > 16) ? 1 : 0;
    constexpr int colT = 4 << ntnsh;             // 4 groups * ntn N-tiles
    int mtiles = (NB << HWsh) >> 4;
    int wid = blockIdx.x * (blockDim.x >> 5) + (threadIdx.x >> 5);
    if (wid >= mtiles * colT) return;            // uniform per wave
    int mt = wid >> (2 + ntnsh);
    int ct = wid & (colT - 1);
    int g  = ct >> ntnsh;
    int nt = ct & ((1 << ntnsh) - 1);

    int lane = threadIdx.x & 31;
    int mrow = lane & 15;
    int half = lane >> 4;
    int ncol = lane & 15;
    int nloc = (nt << 4) + ncol;
    int nlocC = nloc & (Ndim - 1);               // Ndim pow2: safe clamp

    int pbase = mt << 4;
    int nb  = pbase >> HWsh;                     // tile never spans images
    int hwb = pbase & (HW - 1);
    int hw  = hwb + mrow;

    // fold lane-half K offset into bases; per-h offsets are then constants
    const int khoff = (8 * half) & (Kdim - 1);
    size_t abase = ((size_t)nb * inCtot + inC0 + g * Kdim) * HW + hw
                 + (size_t)khoff * HW;
    const float* wbase = w + (size_t)(g * Ndim + nlocC) * Kdim + khoff;

    v16h af, bf;
#pragma unroll
    for (int h = 0; h < 16; ++h) {
        // ISA 16-bit A layout: lanes 0-15 hold K 0-7/16-23, lanes 16-31 hold 8-15/24-31
        const int hbase = (h < 8 ? h : h + 8);
        const int koff  = hbase & (Kdim - 1);    // compile-time constant
        bool kv = (hbase + 8 * half) < Kdim;     // folds for Kdim >= 16
        float av = in[abase + (size_t)koff * HW];
        float wv = wbase[koff];
        af[h] = (_Float16)(kv ? av : 0.f);
        bf[h] = (_Float16)((kv && nloc < Ndim) ? wv : 0.f);
    }
    v8f c = {};
    c = WMMA_F16(false, af, false, bf, (short)0, c, false, false);

    if (nloc < Ndim) {
        int co = g * Ndim + nloc;
        float bv = bias[co];
        float al = alpha[co];
        // lane's 8 D rows are contiguous in hw (base is 32B aligned)
        size_t obase = (((size_t)nb * outCtot + outC0 + co) << HWsh) + hwb + (half << 3);
        float rv[8];
        if (HAS_RES) {
            size_t rbase = (((size_t)nb * resCtot + resC0 + co) << HWsh) + hwb + (half << 3);
            const float4* r4 = (const float4*)(res + rbase);
            float4 rlo = r4[0], rhi = r4[1];
            rv[0] = rlo.x; rv[1] = rlo.y; rv[2] = rlo.z; rv[3] = rlo.w;
            rv[4] = rhi.x; rv[5] = rhi.y; rv[6] = rhi.z; rv[7] = rhi.w;
        }
        float v[8];
#pragma unroll
        for (int r = 0; r < 8; ++r) {
            float t = c[r] + bv;
            if (HAS_RES) t += rv[r];
            v[r] = prelu_f(t, al);
        }
        float4* o4 = (float4*)(out + obase);
        o4[0] = make_float4(v[0], v[1], v[2], v[3]);
        o4[1] = make_float4(v[4], v[5], v[6], v[7]);
    }
}

// ---------------------------------------------------------------------------
// 4 dilated depthwise 3x3 branches + hierarchical fusion + concat + PReLU.
// in: [NB,n,H,W] -> out: [NB,4n,H,W]. Static dims.
// ---------------------------------------------------------------------------
template <int nsh, int Hsh, int Wsh>
__global__ void k_dw_hff(const float* __restrict__ in, const float* __restrict__ dww,
                         const float* __restrict__ dwb, const float* __restrict__ ca,
                         float* __restrict__ out, int NB)
{
    constexpr int n = 1 << nsh, H = 1 << Hsh, W = 1 << Wsh;
    int idx = blockIdx.x * blockDim.x + threadIdx.x;
    int total = NB << (nsh + Hsh + Wsh);
    if (idx >= total) return;
    int xw = idx & (W - 1);
    int t  = idx >> Wsh;
    int y  = t & (H - 1);
    t >>= Hsh;
    int c  = t & (n - 1);
    int nb = t >> nsh;
    constexpr int dil[4] = {1, 1, 2, 3};
    const float* src = in + ((((size_t)nb << nsh) + c) << (Hsh + Wsh));
    float acc = 0.f;
#pragma unroll
    for (int i = 0; i < 4; ++i) {
        constexpr int base_i = 0;  (void)base_i;
        int d = dil[i];
        float sacc = dwb[(i << nsh) + c];
#pragma unroll
        for (int ky = 0; ky < 3; ++ky) {
#pragma unroll
            for (int kx = 0; kx < 3; ++kx) {
                int yy = y + (ky - 1) * d;
                int xx = xw + (kx - 1) * d;
                bool inb = ((unsigned)yy < (unsigned)H) && ((unsigned)xx < (unsigned)W);
                int yyc = inb ? yy : y;
                int xxc = inb ? xx : xw;
                float v = src[(yyc << Wsh) + xxc];
                sacc += dww[(size_t)((i << nsh) + c) * 9 + ky * 3 + kx] * (inb ? v : 0.f);
            }
        }
        acc += sacc;                              // hierarchical fusion
        int oc = (i << nsh) + c;
        out[((((size_t)nb << (nsh + 2)) + oc) << (Hsh + Wsh)) + (y << Wsh) + xw]
            = prelu_f(acc, ca[oc]);
    }
}

// ---------------------------------------------------------------------------
// ECA: per-(n,c) spatial mean (block reduction in LDS). C pow2.
// ---------------------------------------------------------------------------
__global__ void k_mean(const float* __restrict__ in, int inCtot,
                       float* __restrict__ sout, int Csh, int HW)
{
    __shared__ float red[256];
    int nc = blockIdx.x;
    int nb = nc >> Csh; int c = nc & ((1 << Csh) - 1);
    const float* src = in + ((size_t)nb * inCtot + c) * HW;
    float acc = 0.f;
    for (int i = threadIdx.x; i < HW; i += blockDim.x) acc += src[i];
    red[threadIdx.x] = acc;
    __syncthreads();
    for (int st = 128; st > 0; st >>= 1) {
        if ((int)threadIdx.x < st) red[threadIdx.x] += red[threadIdx.x + st];
        __syncthreads();
    }
    if (threadIdx.x == 0) sout[nc] = red[0] / (float)HW;
}

// ECA: 1D conv (k=3) over channels + sigmoid
__global__ void k_gate(const float* __restrict__ sin, const float* __restrict__ w3,
                       float* __restrict__ gate, int NB, int C)
{
    int idx = blockIdx.x * blockDim.x + threadIdx.x;
    if (idx >= NB * C) return;
    int c = idx & (C - 1);
    float y = w3[1] * sin[idx];
    if (c > 0)     y += w3[0] * sin[idx - 1];
    if (c < C - 1) y += w3[2] * sin[idx + 1];
    gate[idx] = 1.f / (1.f + __expf(-y));
}

// ---------------------------------------------------------------------------
// fused ECA-gate * pixel_shuffle(r=2) + PReLU(ps_a). Writes at channel offset
// outC0 into [NB,outCtot,2H,2W]. Static dims, paired b64 stores.
// ---------------------------------------------------------------------------
template <int Cqsh, int Hsh, int Wsh>
__global__ void k_shuffle(const float* __restrict__ in, const float* __restrict__ gate,
                          const float* __restrict__ psa, float* __restrict__ out,
                          int outC0, int outCtot, int NB)
{
    constexpr int W = 1 << Wsh, H = 1 << Hsh;
    int idx = blockIdx.x * blockDim.x + threadIdx.x;
    int total = NB << (Cqsh + Hsh + Wsh);
    if (idx >= total) return;
    int xw = idx & (W - 1);
    int t  = idx >> Wsh;
    int y  = t & (H - 1);
    t >>= Hsh;
    int cq = t & ((1 << Cqsh) - 1);
    int nb = t >> Cqsh;
    float al = psa[cq];
    size_t ib = (((size_t)nb << (Cqsh + 2)) + (cq << 2)) << (Hsh + Wsh);
    int    gb = (nb << (Cqsh + 2)) + (cq << 2);
    size_t obase = ((size_t)nb * outCtot + outC0 + cq) << (Hsh + Wsh + 2);
#pragma unroll
    for (int i = 0; i < 2; ++i) {
        float v0 = in[ib + ((size_t)(i * 2 + 0) << (Hsh + Wsh)) + (y << Wsh) + xw]
                   * gate[gb + i * 2 + 0];
        float v1 = in[ib + ((size_t)(i * 2 + 1) << (Hsh + Wsh)) + (y << Wsh) + xw]
                   * gate[gb + i * 2 + 1];
        float2 o;
        o.x = prelu_f(v0, al);
        o.y = prelu_f(v1, al);
        *(float2*)(out + obase + ((size_t)(2 * y + i) << (Wsh + 1)) + 2 * xw) = o;
    }
}

// ---------------------------------------------------------------------------
// 3x3 conv as implicit WMMA GEMM: Cin=32 fixed, Cout<=16, static dims.
// 9 taps x one K=32 WMMA. Used for dec (32->1).
// ---------------------------------------------------------------------------
template <int Cout, int Hsh, int Wsh>
__global__ void k_conv3_wmma(const float* __restrict__ in, const float* __restrict__ w,
                             const float* __restrict__ bias, const float* __restrict__ alpha,
                             float* __restrict__ out)
{
    constexpr int Cin = 32;
    constexpr int H = 1 << Hsh, W = 1 << Wsh;
    constexpr int HWsh = Hsh + Wsh, HW = 1 << HWsh;
    int wid = blockIdx.x * (blockDim.x >> 5) + (threadIdx.x >> 5);
    int lane = threadIdx.x & 31;
    int mrow = lane & 15, half = lane >> 4, ncol = lane & 15;
    int pbase = wid << 4;
    int nb  = pbase >> HWsh;
    int hwb = pbase & (HW - 1);
    int hw  = hwb + mrow;
    int y = hw >> Wsh, xw = hw & (W - 1);
    size_t cb = (((size_t)nb * Cin) << HWsh) + ((size_t)(8 * half) << HWsh);
    int ncolC = ncol < Cout ? ncol : 0;
    const float* wb = w + (size_t)ncolC * Cin * 9 + (size_t)(8 * half) * 9;
    v8f c = {};
#pragma unroll
    for (int t = 0; t < 9; ++t) {
        const int kh = t / 3, kw = t % 3;
        int yy = y + kh - 1, xx = xw + kw - 1;
        bool inb = ((unsigned)yy < (unsigned)H) && ((unsigned)xx < (unsigned)W);
        int yyc = inb ? yy : y;
        int xxc = inb ? xx : xw;
        size_t base = cb + ((size_t)yyc << Wsh) + xxc;
        v16h af, bf;
#pragma unroll
        for (int h = 0; h < 16; ++h) {
            const int hbase = (h < 8 ? h : h + 8);   // + 8*half folded into bases
            float av = in[base + ((size_t)hbase << HWsh)];
            af[h] = (_Float16)(inb ? av : 0.f);
            float wv = wb[(size_t)hbase * 9 + kh * 3 + kw];
            bf[h] = (_Float16)((ncol < Cout) ? wv : 0.f);
        }
        c = WMMA_F16(false, af, false, bf, (short)0, c, false, false);
    }
    if (ncol < Cout) {
        float bv = bias[ncol]; float al = alpha[ncol];
        size_t ob = (((size_t)nb * Cout + ncol) << HWsh) + hwb + (half << 3);
        float v[8];
#pragma unroll
        for (int r = 0; r < 8; ++r) v[r] = prelu_f(c[r] + bv, al);
        float4* o4 = (float4*)(out + ob);
        o4[0] = make_float4(v[0], v[1], v[2], v[3]);
        o4[1] = make_float4(v[4], v[5], v[6], v[7]);
    }
}

// ---------------------------------------------------------------------------
// fused: z4 = relu(conv3x3(up22, k4_w)) via WMMA, staged to LDS, then
// out = dec + sum_j decPad(y+kh,x+kw) * z4[j],  j = kw*3 + kh. Exact grid.
// ---------------------------------------------------------------------------
template <int Hsh, int Wsh>
__global__ void k_k4pix(const float* __restrict__ in, const float* __restrict__ w,
                        const float* __restrict__ bias, const float* __restrict__ dec,
                        float* __restrict__ outp)
{
    constexpr int Cin = 32, Cout = 9;
    constexpr int H = 1 << Hsh, W = 1 << Wsh;
    constexpr int HWsh = Hsh + Wsh, HW = 1 << HWsh;
    __shared__ float lds[4][16][12];
    int wslot = threadIdx.x >> 5;
    int wid = blockIdx.x * (blockDim.x >> 5) + wslot;
    int lane = threadIdx.x & 31;
    int mrow = lane & 15, half = lane >> 4, ncol = lane & 15;
    int pbase = wid << 4;
    int nb  = pbase >> HWsh;
    int hwb = pbase & (HW - 1);
    int hw  = hwb + mrow;
    int y = hw >> Wsh, xw = hw & (W - 1);
    size_t cb = (((size_t)nb * Cin) << HWsh) + ((size_t)(8 * half) << HWsh);
    int ncolC = ncol < Cout ? ncol : 0;
    const float* wb = w + (size_t)ncolC * Cin * 9 + (size_t)(8 * half) * 9;
    v8f c = {};
#pragma unroll
    for (int t = 0; t < 9; ++t) {
        const int kh = t / 3, kw = t % 3;
        int yy = y + kh - 1, xx = xw + kw - 1;
        bool inb = ((unsigned)yy < (unsigned)H) && ((unsigned)xx < (unsigned)W);
        int yyc = inb ? yy : y;
        int xxc = inb ? xx : xw;
        size_t base = cb + ((size_t)yyc << Wsh) + xxc;
        v16h af, bf;
#pragma unroll
        for (int h = 0; h < 16; ++h) {
            const int hbase = (h < 8 ? h : h + 8);
            float av = in[base + ((size_t)hbase << HWsh)];
            af[h] = (_Float16)(inb ? av : 0.f);
            float wv = wb[(size_t)hbase * 9 + kh * 3 + kw];
            bf[h] = (_Float16)((ncol < Cout) ? wv : 0.f);
        }
        c = WMMA_F16(false, af, false, bf, (short)0, c, false, false);
    }
    if (ncol < Cout) {
#pragma unroll
        for (int r = 0; r < 8; ++r) {
            float v = c[r] + bias[ncol];
            lds[wslot][r + half * 8][ncol] = v > 0.f ? v : 0.f;   // ReLU(k4)
        }
    }
    __syncthreads();
    if (lane < 16) {
        int hw2 = hwb + lane;
        int y2 = hw2 >> Wsh, x2 = hw2 & (W - 1);
        const float* dsl = dec + ((size_t)nb << HWsh);
        float z = 0.f;
#pragma unroll
        for (int kw2 = 0; kw2 < 3; ++kw2)
#pragma unroll
            for (int kh2 = 0; kh2 < 3; ++kh2) {
                int j = kw2 * 3 + kh2;
                int yy = y2 + kh2 - 1, xx = x2 + kw2 - 1;
                bool inb = ((unsigned)yy < (unsigned)H) && ((unsigned)xx < (unsigned)W);
                int yyc = inb ? yy : y2;
                int xxc = inb ? xx : x2;
                float dv = dsl[(yyc << Wsh) + xxc];
                z += (inb ? dv : 0.f) * lds[wslot][lane][j];
            }
        outp[((size_t)nb << HWsh) + hw2] = dsl[(y2 << Wsh) + x2] + z;   // out + z4
    }
}

// ===========================================================================
// host orchestration (fully static shape propagation via templates)
// ===========================================================================
struct EespP { const float *pw, *pb, *pa, *dww, *dwb, *ca, *ew, *eb, *aa; };

template <int HWsh, int Kdim, int Ndim>
static void launch_pw(const float* in, int inC0, int inCtot,
                      const float* w, const float* bias, const float* alpha,
                      const float* res, int resC0, int resCtot,
                      float* out, int outC0, int outCtot, int NB, hipStream_t s)
{
    constexpr int ntnsh = (Ndim > 16) ? 1 : 0;
    int mtiles = (NB << HWsh) >> 4;
    int blocks = (mtiles << (2 + ntnsh)) / 4;    // waves always divisible by 4
    if (res)
        k_pw_wmma<true, HWsh, Kdim, Ndim><<<blocks, 128, 0, s>>>(
            in, inC0, inCtot, w, bias, alpha, res, resC0, resCtot,
            out, outC0, outCtot, NB);
    else
        k_pw_wmma<false, HWsh, Kdim, Ndim><<<blocks, 128, 0, s>>>(
            in, inC0, inCtot, w, bias, alpha, nullptr, 0, 1,
            out, outC0, outCtot, NB);
}

template <int Hsh, int Wsh, int Cin, int Cout>
static void run_eesp(const float* in, int inC0, int inCtot, const EespP& E, bool residual,
                     float* out, int outC0, int outCtot,
                     float* T1, float* T2, int NB, hipStream_t s)
{
    constexpr int n = Cout / 4;
    constexpr int HWsh = Hsh + Wsh;
    // grouped 1x1 proj + PReLU  (Cin -> n)
    launch_pw<HWsh, Cin / 4, n / 4>(in, inC0, inCtot, E.pw, E.pb, E.pa,
                                    nullptr, 0, 1, T1, 0, n, NB, s);
    // 4 dilated depthwise branches + HFF + concat + PReLU  (n -> 4n)
    constexpr int nsh = clog2(n);
    int total = NB << (nsh + HWsh);
    k_dw_hff<nsh, Hsh, Wsh><<<divUp(total, 256), 256, 0, s>>>(T1, E.dww, E.dwb, E.ca,
                                                              T2, NB);
    // grouped 1x1 expand (+ residual) + PReLU  (4n -> 4n)
    launch_pw<HWsh, n, n>(T2, 0, 4 * n, E.ew, E.eb, E.aa,
                          residual ? in : nullptr, inC0, inCtot,
                          out, outC0, outCtot, NB, s);
}

template <int Hsh, int Wsh, int Cin, int Cout>
static void run_branch(const float* in, int inC0, int inCtot, const EespP& E,
                       const float* sew, const float* psa, bool residual,
                       float* out, int outC0, int outCtot,
                       float* T1, float* T2, float* T3, float* SB, float* GB,
                       int NB, hipStream_t s)
{
    run_eesp<Hsh, Wsh, Cin, Cout>(in, inC0, inCtot, E, residual, T3, 0, Cout,
                                  T1, T2, NB, s);
    k_mean<<<NB * Cout, 256, 0, s>>>(T3, Cout, SB, clog2(Cout), 1 << (Hsh + Wsh));
    k_gate<<<divUp(NB * Cout, 256), 256, 0, s>>>(SB, sew, GB, NB, Cout);
    constexpr int Cqsh = clog2(Cout) - 2;
    int total = NB << (Cqsh + Hsh + Wsh);
    k_shuffle<Cqsh, Hsh, Wsh><<<divUp(total, 256), 256, 0, s>>>(T3, GB, psa,
                                                                out, outC0, outCtot, NB);
}

extern "C" void kernel_launch(void* const* d_in, const int* in_sizes, int n_in,
                              void* d_out, int out_size, void* d_ws, size_t ws_size,
                              hipStream_t stream)
{
    (void)in_sizes; (void)n_in; (void)out_size; (void)ws_size;
    const float* const* P = (const float* const*)d_in;
    // setup_inputs dict order, params flattened depth-first in insertion order
    const float* x      = P[0];
    const float* head_w = P[1]; const float* head_b = P[2]; const float* head_a = P[3];
    EespP eM   = {P[4],  P[5],  P[6],  P[7],  P[8],  P[9],  P[10], P[11], P[12]};
    EespP e121 = {P[13], P[14], P[15], P[16], P[17], P[18], P[19], P[20], P[21]};
    const float* se121 = P[22]; const float* ps121 = P[23];
    EespP e122 = {P[24], P[25], P[26], P[27], P[28], P[29], P[30], P[31], P[32]};
    const float* se122 = P[33]; const float* ps122 = P[34];
    EespP e241 = {P[35], P[36], P[37], P[38], P[39], P[40], P[41], P[42], P[43]};
    const float* se241 = P[44]; const float* ps241 = P[45];
    EespP e242 = {P[46], P[47], P[48], P[49], P[50], P[51], P[52], P[53], P[54]};
    const float* se242 = P[55]; const float* ps242 = P[56];
    const float* dec_w = P[57]; const float* dec_b = P[58]; const float* dec_a = P[59];
    const float* k4_w  = P[60]; const float* k4_b  = P[61];

    // workspace layout (fp32 elements)
    float* ws   = (float*)d_ws;
    float* HEAD = ws;                    //  8*64*128*128 =  8388608
    float* X1   = HEAD + 8388608;        //  8388608
    float* T1   = X1   + 8388608;        //  8388608 (proj outs, max 8*16*256*256)
    float* T2   = T1   + 8388608;        // 33554432 (cat, max 8*64*256*256)
    float* T3   = T2   + 33554432;       // 33554432 (eesp out)
    float* UP12 = T3   + 33554432;       // 33554432 = [8,64,256,256]
    float* UP22 = UP12 + 33554432;       // 67108864 = [8,32,512,512]
    float* DEC  = UP22 + 67108864;       //  2097152 = [8,1,512,512]
    float* SB   = DEC  + 2097152;        //  1024 (channel means)
    float* GB   = SB   + 1024;           //  1024 (ECA gates)

    const int NB = 8;

    // head: 1 -> 64 @128x128
    {
        int total = NB << (6 + 7 + 7);
        k_head<6, 7, 7><<<divUp(total, 256), 256, 0, stream>>>(x, head_w, head_b, head_a,
                                                               HEAD, NB);
    }
    // up11 = branch(head)            -> UP12 channels [0,32)
    run_branch<7, 7, 64, 128>(HEAD, 0, 64, e121, se121, ps121, false, UP12, 0, 64,
                              T1, T2, T3, SB, GB, NB, stream);
    // x1 = eesp(head, residual)
    run_eesp<7, 7, 64, 64>(HEAD, 0, 64, eM, true, X1, 0, 64, T1, T2, NB, stream);
    // up12 = branch(x1)              -> UP12 channels [32,64)
    run_branch<7, 7, 64, 128>(X1, 0, 64, e122, se122, ps122, false, UP12, 32, 64,
                              T1, T2, T3, SB, GB, NB, stream);
    // up21 = branch(up11)            -> UP22 channels [0,16)
    run_branch<8, 8, 32, 64>(UP12, 0, 64, e241, se241, ps241, false, UP22, 0, 32,
                             T1, T2, T3, SB, GB, NB, stream);
    // up22 = branch(up12, residual)  -> UP22 channels [16,32)
    run_branch<8, 8, 64, 64>(UP12, 0, 64, e242, se242, ps242, true, UP22, 16, 32,
                             T1, T2, T3, SB, GB, NB, stream);
    // dec: 32 -> 1 @512x512 (implicit-GEMM WMMA), PReLU
    {
        int waves = (NB << 18) >> 4;     // 131072, divisible by 4
        k_conv3_wmma<1, 9, 9><<<waves / 4, 128, 0, stream>>>(UP22, dec_w, dec_b, dec_a,
                                                             DEC);
    }
    // z4 = relu(conv 32->9) fused with pixel_conv; out = dec + z4
    {
        int waves = (NB << 18) >> 4;
        k_k4pix<9, 9><<<waves / 4, 128, 0, stream>>>(UP22, k4_w, k4_b, DEC,
                                                     (float*)d_out);
    }
}
